// GNN_19868518711704
// MI455X (gfx1250) — matrix operand order repaired and checked by reading, not compiled, not run on hardware
//
#include <hip/hip_runtime.h>

// MI455X / gfx1250, wave32. Compile-only target.

typedef __attribute__((ext_vector_type(16))) _Float16 v16h;
typedef __attribute__((ext_vector_type(8)))  float    v8f;

#define N_NODES 50000
#define E_EDGES 800000
#define C_DIM   128
#define G_DIM   128
#define L0_DIM  512
#define L1_DIM  256

// ---------- helpers ----------
__device__ __forceinline__ unsigned fenc(float f) {
  unsigned u = __float_as_uint(f);
  return (u & 0x80000000u) ? ~u : (u | 0x80000000u);
}
__device__ __forceinline__ float fdec(unsigned u) {
  return (u & 0x80000000u) ? __uint_as_float(u & 0x7fffffffu)
                           : __uint_as_float(~u);
}
__device__ __forceinline__ void atomAddF(float* p, float v) {
  __hip_atomic_fetch_add(p, v, __ATOMIC_RELAXED, __HIP_MEMORY_SCOPE_AGENT);
}
__device__ __forceinline__ void atomMaxU(unsigned* p, unsigned v) {
  __hip_atomic_fetch_max(p, v, __ATOMIC_RELAXED, __HIP_MEMORY_SCOPE_AGENT);
}

// ---------- generic zero ----------
__global__ void zero_kernel(unsigned* __restrict__ p, int n) {
  int t = blockIdx.x * blockDim.x + threadIdx.x;
  if (t < n) p[t] = 0u;
}

// ---------- pack W[K][N] (f32 row-major) into WMMA B-fragment order (f16) ----------
// B fragment (32x16, f16): lane l -> n = l&15, k-group = l>>4; element e -> k = kt*32 + ((e&8)?16:0) + (l>>4)*8 + (e&7)
__global__ void pack_b_kernel(const float* __restrict__ W, _Float16* __restrict__ Bp,
                              int K, int N, int ntile_off) {
  int ktiles = K >> 5;
  int total  = (N >> 4) * ktiles * 512;
  int t = blockIdx.x * blockDim.x + threadIdx.x;
  if (t >= total) return;
  int e    = t & 15;
  int lane = (t >> 4) & 31;
  int kt   = (t >> 9) % ktiles;
  int nt   = (t >> 9) / ktiles;
  int k = kt * 32 + ((e & 8) ? 16 : 0) + (lane >> 4) * 8 + (e & 7);
  int n = nt * 16 + (lane & 15);
  Bp[(((size_t)(ntile_off + nt) * ktiles + kt) * 32 + lane) * 16 + e] =
      (_Float16)W[(size_t)k * N + n];
}

__global__ void concat_bias_kernel(const float* __restrict__ b0, const float* __restrict__ b1,
                                   const float* __restrict__ b2, const float* __restrict__ b3,
                                   float* __restrict__ out) {
  int t = blockIdx.x * blockDim.x + threadIdx.x;
  if (t >= 512) return;
  const float* src = (t < 128) ? b0 : (t < 256) ? b1 : (t < 384) ? b2 : b3;
  out[t] = src[t & 127];
}

// ---------- WMMA GEMM: C[M][Nout] = act(A[M][K] @ Bpacked + bias) ----------
// grid = (M/16, Nout/128), block = 256 (8 waves, one 16x16 tile each)
__global__ __launch_bounds__(256)
void wmma_gemm_kernel(const float* __restrict__ A, int K,
                      const _Float16* __restrict__ Bp,
                      const float* __restrict__ bias, int Nout,
                      float* __restrict__ Cout, int act) {
  extern __shared__ __align__(16) _Float16 Alds[];   // [16][K+8]
  const int KP     = K + 8;
  const int mtile  = blockIdx.x;
  const int wave   = threadIdx.x >> 5;
  const int lane   = threadIdx.x & 31;
  const int ntile  = blockIdx.y * 8 + wave;
  const int ktiles = K >> 5;

  // stage A tile 16 x K, convert fp32 -> f16 in LDS
  for (int i = threadIdx.x; i < 16 * K; i += 256) {
    int r = i / K, c = i - r * K;
    Alds[r * KP + c] = (_Float16)A[(size_t)(mtile * 16 + r) * K + c];
  }
  __syncthreads();

  const int row  = lane & 15;
  const int half = lane >> 4;
  v8f acc = {};
  for (int kt = 0; kt < ktiles; ++kt) {
    union { v16h h; unsigned u[8]; } a;
#pragma unroll
    for (int p = 0; p < 8; ++p) {
      // A 16x32 f16 layout (ISA 7.12.2): VGPR p, lane half -> K base
      int k = kt * 32 + ((p & 4) ? 16 : 0) + half * 8 + 2 * (p & 3);
      a.u[p] = *(const unsigned*)&Alds[row * KP + k];
    }
    v16h b = *(const v16h*)(Bp + (((size_t)ntile * ktiles + kt) * 32 + lane) * 16);
    acc = __builtin_amdgcn_wmma_f32_16x16x32_f16(false, a.h, false, b,
                                                 (short)0, acc, false, false);
  }

  const int col = ntile * 16 + (lane & 15);
  const float bv = bias[col];
#pragma unroll
  for (int r = 0; r < 8; ++r) {
    int mrow = mtile * 16 + half * 8 + r;   // C/D layout: VGPR r -> M = r + 8*half
    float v = acc[r] + bv;
    if (act == 1)      v = fmaxf(v, 0.0f);
    else if (act == 2) v = (v > 0.0f) ? v : 0.01f * v;
    Cout[(size_t)mrow * Nout + col] = v;
  }
}

// ---------- layer 0 projections (din = 4): Z = x@[Wq|Wk|Wv|Ws] + b ----------
__global__ __launch_bounds__(128)
void layer0_kernel(const float* __restrict__ x,
                   const float* __restrict__ Wq, const float* __restrict__ bq,
                   const float* __restrict__ Wk, const float* __restrict__ bk,
                   const float* __restrict__ Wv, const float* __restrict__ bv,
                   const float* __restrict__ Ws, const float* __restrict__ bs,
                   float* __restrict__ Z, int N) {
  int n = blockIdx.x;
  int c = threadIdx.x;
  if (n >= N) return;
  float x0 = x[n * 4], x1 = x[n * 4 + 1], x2 = x[n * 4 + 2], x3 = x[n * 4 + 3];
  float q = bq[c] + x0 * Wq[c] + x1 * Wq[128 + c] + x2 * Wq[256 + c] + x3 * Wq[384 + c];
  float k = bk[c] + x0 * Wk[c] + x1 * Wk[128 + c] + x2 * Wk[256 + c] + x3 * Wk[384 + c];
  float v = bv[c] + x0 * Wv[c] + x1 * Wv[128 + c] + x2 * Wv[256 + c] + x3 * Wv[384 + c];
  float s = bs[c] + x0 * Ws[c] + x1 * Ws[128 + c] + x2 * Ws[256 + c] + x3 * Ws[384 + c];
  float* z = Z + (size_t)n * 512;
  z[c] = q; z[128 + c] = k; z[256 + c] = v; z[384 + c] = s;
}

// ---------- edge phase A: logits + segment max (wave per edge) ----------
__global__ __launch_bounds__(256)
void edge_logits_kernel(const int* __restrict__ ei, const float* __restrict__ eattr,
                        const float* __restrict__ We, const float* __restrict__ Z,
                        float* __restrict__ aw, unsigned* __restrict__ menc, int E) {
  __shared__ float WeS[384];
  for (int i = threadIdx.x; i < 384; i += 256) WeS[i] = We[i];
  __syncthreads();
  int eid  = blockIdx.x * 8 + (threadIdx.x >> 5);
  int lane = threadIdx.x & 31;
  if (eid >= E) return;
  int s = ei[eid], d = ei[E + eid];
  float ea0 = eattr[eid * 3], ea1 = eattr[eid * 3 + 1], ea2 = eattr[eid * 3 + 2];
  const float* q  = Z + (size_t)d * 512;
  const float* kk = Z + (size_t)s * 512 + 128;
  float dot = 0.0f;
#pragma unroll
  for (int j = 0; j < 4; ++j) {
    int c = lane + j * 32;
    float ec = ea0 * WeS[c] + ea1 * WeS[128 + c] + ea2 * WeS[256 + c];
    dot += q[c] * (kk[c] + ec);
  }
#pragma unroll
  for (int off = 16; off; off >>= 1) dot += __shfl_xor(dot, off, 32);
  dot *= 0.088388347648318447f;   // 1/sqrt(128)
  if (lane == 0) {
    aw[eid] = dot;
    atomMaxU(menc + d, fenc(dot));
  }
}

// ---------- edge phase B: exp + segment sum (thread per edge) ----------
__global__ void edge_exp_kernel(const int* __restrict__ ei, float* __restrict__ aw,
                                const unsigned* __restrict__ menc,
                                float* __restrict__ den, int E) {
  int e = blockIdx.x * blockDim.x + threadIdx.x;
  if (e >= E) return;
  int d = ei[E + e];
  float a = __expf(aw[e] - fdec(menc[d]));
  aw[e] = a;
  atomAddF(den + d, a);
}

// ---------- edge phase C: weighted scatter into skip columns of Z ----------
__global__ __launch_bounds__(256)
void edge_scatter_kernel(const int* __restrict__ ei, const float* __restrict__ eattr,
                         const float* __restrict__ We, const float* __restrict__ aw,
                         const float* __restrict__ den, float* __restrict__ Z, int E) {
  __shared__ float WeS[384];
  for (int i = threadIdx.x; i < 384; i += 256) WeS[i] = We[i];
  __syncthreads();
  int eid  = blockIdx.x * 8 + (threadIdx.x >> 5);
  int lane = threadIdx.x & 31;
  if (eid >= E) return;
  int s = ei[eid], d = ei[E + eid];
  float alpha = aw[eid] / den[d];
  float ea0 = eattr[eid * 3], ea1 = eattr[eid * 3 + 1], ea2 = eattr[eid * 3 + 2];
  const float* vv = Z + (size_t)s * 512 + 256;
  float* out = Z + (size_t)d * 512 + 384;
#pragma unroll
  for (int j = 0; j < 4; ++j) {
    int c = lane + j * 32;
    float ec = ea0 * WeS[c] + ea1 * WeS[128 + c] + ea2 * WeS[256 + c];
    atomAddF(out + c, (vv[c] + ec) * alpha);
  }
}

// ---------- leaky-relu Z[:,384:512] -> h ----------
__global__ void finish_kernel(const float* __restrict__ Z, float* __restrict__ h, int N) {
  int t = blockIdx.x * blockDim.x + threadIdx.x;
  if (t >= N * C_DIM) return;
  int n = t >> 7, c = t & 127;
  float v = Z[(size_t)n * 512 + 384 + c];
  h[t] = (v > 0.0f) ? v : 0.01f * v;
}

// ---------- leaky-relu + global_add_pool ----------
__global__ void pool_kernel(const float* __restrict__ Z, const int* __restrict__ batch,
                            float* __restrict__ g, int N) {
  int t = blockIdx.x * blockDim.x + threadIdx.x;
  if (t >= N * C_DIM) return;
  int n = t >> 7, c = t & 127;
  float v = Z[(size_t)n * 512 + 384 + c];
  v = (v > 0.0f) ? v : 0.01f * v;
  atomAddF(g + (size_t)batch[n] * C_DIM + c, v);
}

// ---------- final 256 -> 2 ----------
__global__ void mlp_final_kernel(const float* __restrict__ g2, const float* __restrict__ W,
                                 const float* __restrict__ b, float* __restrict__ out) {
  int t = blockIdx.x * blockDim.x + threadIdx.x;
  if (t >= G_DIM * 2) return;
  int gi = t >> 1, p = t & 1;
  float acc = b[p];
  for (int c = 0; c < L1_DIM; ++c) acc += g2[(size_t)gi * L1_DIM + c] * W[c * 2 + p];
  out[t] = acc;
}

// =====================================================================
extern "C" void kernel_launch(void* const* d_in, const int* in_sizes, int n_in,
                              void* d_out, int out_size, void* d_ws, size_t ws_size,
                              hipStream_t stream) {
  const float* x     = (const float*)d_in[0];
  const int*   ei    = (const int*)  d_in[1];
  const float* eattr = (const float*)d_in[2];
  const int*   batch = (const int*)  d_in[3];
  const float *Wq0=(const float*)d_in[4],  *bq0=(const float*)d_in[5];
  const float *Wk0=(const float*)d_in[6],  *bk0=(const float*)d_in[7];
  const float *Wv0=(const float*)d_in[8],  *bv0=(const float*)d_in[9];
  const float *We0=(const float*)d_in[10], *Ws0=(const float*)d_in[11], *bs0=(const float*)d_in[12];
  const float *Wq1=(const float*)d_in[13], *bq1=(const float*)d_in[14];
  const float *Wk1=(const float*)d_in[15], *bk1=(const float*)d_in[16];
  const float *Wv1=(const float*)d_in[17], *bv1=(const float*)d_in[18];
  const float *We1=(const float*)d_in[19], *Ws1=(const float*)d_in[20], *bs1=(const float*)d_in[21];
  const float *Wl0=(const float*)d_in[22], *bl0=(const float*)d_in[23];
  const float *Wl1=(const float*)d_in[24], *bl1=(const float*)d_in[25];
  const float *Wl2=(const float*)d_in[26], *bl2=(const float*)d_in[27];

  const int N = N_NODES, E = E_EDGES;

  // workspace carve-out (256B aligned)
  char* ws = (char*)d_ws;
  size_t off = 0;
  auto carve = [&](size_t bytes) -> char* {
    char* p = ws + off;
    off = (off + bytes + 255) & ~(size_t)255;
    return p;
  };
  float*     Z     = (float*)    carve((size_t)N * 512 * 4);   // q|k|v|acc
  float*     h     = (float*)    carve((size_t)N * C_DIM * 4); // layer activations
  float*     aw    = (float*)    carve((size_t)E * 4);         // logits -> exp weights
  unsigned*  menc  = (unsigned*) carve((size_t)N * 4);
  float*     den   = (float*)    carve((size_t)N * 4);
  float*     g     = (float*)    carve((size_t)G_DIM * C_DIM * 4);
  float*     g1    = (float*)    carve((size_t)G_DIM * L0_DIM * 4);
  float*     g2    = (float*)    carve((size_t)G_DIM * L1_DIM * 4);
  _Float16*  Bqkvs = (_Float16*) carve((size_t)32 * 4  * 512 * 2); // K=128,Nout=512
  float*     bqkvs = (float*)    carve(512 * 4);
  _Float16*  Bh0   = (_Float16*) carve((size_t)32 * 4  * 512 * 2); // K=128,Nout=512
  _Float16*  Bh1   = (_Float16*) carve((size_t)16 * 16 * 512 * 2); // K=512,Nout=256
  (void)ws_size; (void)in_sizes; (void)n_in; (void)out_size;

  // ---- pack weights (deterministic; every call) ----
  {
    int tot = (C_DIM / 16) * (C_DIM / 32) * 512;         // 16384 per matrix
    int blk = (tot + 255) / 256;
    pack_b_kernel<<<blk, 256, 0, stream>>>(Wq1, Bqkvs, C_DIM, C_DIM, 0);
    pack_b_kernel<<<blk, 256, 0, stream>>>(Wk1, Bqkvs, C_DIM, C_DIM, 8);
    pack_b_kernel<<<blk, 256, 0, stream>>>(Wv1, Bqkvs, C_DIM, C_DIM, 16);
    pack_b_kernel<<<blk, 256, 0, stream>>>(Ws1, Bqkvs, C_DIM, C_DIM, 24);
    concat_bias_kernel<<<2, 256, 0, stream>>>(bq1, bk1, bv1, bs1, bqkvs);
    int tot0 = (L0_DIM / 16) * (C_DIM / 32) * 512;       // 65536
    pack_b_kernel<<<(tot0 + 255) / 256, 256, 0, stream>>>(Wl0, Bh0, C_DIM, L0_DIM, 0);
    int tot1 = (L1_DIM / 16) * (L0_DIM / 32) * 512;      // 131072
    pack_b_kernel<<<(tot1 + 255) / 256, 256, 0, stream>>>(Wl1, Bh1, L0_DIM, L1_DIM, 0);
  }

  const int zeroN = (N + 255) / 256;
  const int egrid = E / 8;                   // 800000 % 8 == 0
  const int eth   = (E + 255) / 256;
  const int nct   = (N * C_DIM) / 256;       // 25000

  // ================= layer 0 (din = 4) =================
  layer0_kernel<<<N, 128, 0, stream>>>(x, Wq0, bq0, Wk0, bk0, Wv0, bv0, Ws0, bs0, Z, N);
  zero_kernel<<<zeroN, 256, 0, stream>>>(menc, N);
  zero_kernel<<<zeroN, 256, 0, stream>>>((unsigned*)den, N);
  edge_logits_kernel <<<egrid, 256, 0, stream>>>(ei, eattr, We0, Z, aw, menc, E);
  edge_exp_kernel    <<<eth,   256, 0, stream>>>(ei, aw, menc, den, E);
  edge_scatter_kernel<<<egrid, 256, 0, stream>>>(ei, eattr, We0, aw, den, Z, E);
  finish_kernel<<<nct, 256, 0, stream>>>(Z, h, N);

  // ================= layers 1 & 2 (shared weights) =================
  const size_t shA128 = (size_t)16 * (C_DIM + 8) * sizeof(_Float16);
  for (int layer = 1; layer <= 2; ++layer) {
    wmma_gemm_kernel<<<dim3(N / 16, 512 / 128), 256, shA128, stream>>>(
        h, C_DIM, Bqkvs, bqkvs, 512, Z, /*act=*/0);
    zero_kernel<<<zeroN, 256, 0, stream>>>(menc, N);
    zero_kernel<<<zeroN, 256, 0, stream>>>((unsigned*)den, N);
    edge_logits_kernel <<<egrid, 256, 0, stream>>>(ei, eattr, We1, Z, aw, menc, E);
    edge_exp_kernel    <<<eth,   256, 0, stream>>>(ei, aw, menc, den, E);
    edge_scatter_kernel<<<egrid, 256, 0, stream>>>(ei, eattr, We1, aw, den, Z, E);
    if (layer == 1) {
      finish_kernel<<<nct, 256, 0, stream>>>(Z, h, N);
    } else {
      zero_kernel<<<(G_DIM * C_DIM + 255) / 256, 256, 0, stream>>>((unsigned*)g, G_DIM * C_DIM);
      pool_kernel<<<nct, 256, 0, stream>>>(Z, batch, g, N);
    }
  }

  // ================= MLP head =================
  wmma_gemm_kernel<<<dim3(G_DIM / 16, L0_DIM / 128), 256, shA128, stream>>>(
      g, C_DIM, Bh0, bl0, L0_DIM, g1, /*act=*/1);
  const size_t shA512 = (size_t)16 * (L0_DIM + 8) * sizeof(_Float16);
  wmma_gemm_kernel<<<dim3(G_DIM / 16, L1_DIM / 128), 256, shA512, stream>>>(
      g1, L0_DIM, Bh1, bl1, L1_DIM, g2, /*act=*/1);
  mlp_final_kernel<<<1, 256, 0, stream>>>(g2, Wl2, bl2, (float*)d_out);
}
